// GVCNN_68315749810578
// MI455X (gfx1250) — matrix operand order repaired and checked by reading, not compiled, not run on hardware
//
#include <hip/hip_runtime.h>
#include <hip/hip_bf16.h>
#include <stddef.h>

// CDNA5 (gfx1250) wave32 WMMA types
typedef __attribute__((ext_vector_type(16))) _Float16 v16h;
typedef __attribute__((ext_vector_type(8)))  float    v8f;

// ---------------------------------------------------------------------------
// elementwise / prep kernels
// ---------------------------------------------------------------------------
__global__ void k_f32_to_f16(const float* __restrict__ src, _Float16* __restrict__ dst, size_t n) {
    size_t i = (size_t)blockIdx.x * blockDim.x + threadIdx.x;
    if (i < n) dst[i] = (_Float16)src[i];
}

__global__ void k_zero_f32(float* __restrict__ dst, int n) {
    int i = blockIdx.x * blockDim.x + threadIdx.x;
    if (i < n) dst[i] = 0.f;
}

// fold BN into per-channel scale/shift: y = x*scale + shift
__global__ void k_bn_fuse(const float* __restrict__ g, const float* __restrict__ b,
                          const float* __restrict__ m, const float* __restrict__ v,
                          float* __restrict__ scale, float* __restrict__ shift, int C) {
    int i = blockIdx.x * blockDim.x + threadIdx.x;
    if (i < C) {
        float s = g[i] * rsqrtf(v[i] + 1e-5f);
        scale[i] = s;
        shift[i] = b[i] - m[i] * s;
    }
}

// Pack OIHW fp32 conv weights into this kernel's WMMA A-fragment layout (f16, contiguous k order):
//   dst[((tileCo*nKc + kc)*32 + lane)*16 + h]
//   K within a 32-chunk:  K = ((lane&16)?16:0) + h
// so the conv kernel's A-fragment load is one contiguous 32B vector per lane.
__global__ void k_pack_w(const float* __restrict__ w, _Float16* __restrict__ dst,
                         int Cout, int Kreal, int Kpad) {
    int nKc = Kpad >> 5;
    size_t total = (size_t)(Cout >> 4) * (size_t)nKc * 512;
    size_t idx = (size_t)blockIdx.x * blockDim.x + threadIdx.x;
    if (idx >= total) return;
    int h    = (int)(idx & 15);
    int lane = (int)((idx >> 4) & 31);
    size_t t = idx >> 9;
    int kc     = (int)(t % nKc);
    int tileCo = (int)(t / nKc);
    int co = (tileCo << 4) + (lane & 15);
    int k  = (kc << 5) + ((lane & 16) ? 16 : 0) + h;
    float val = (k < Kreal) ? w[(size_t)co * Kreal + k] : 0.f;
    dst[idx] = (_Float16)val;
}

// ---------------------------------------------------------------------------
// Implicit-GEMM conv via v_wmma_f32_16x16x32_f16, with block-shared im2col.
//
// All waves of a block share one 16-pixel tile and handle consecutive Cout
// tiles. The 32-K B chunk (512 f16 = 1KB) is gathered cooperatively into LDS
// in WMMA fragment order (double-buffered, one barrier per chunk); every wave
// then reads its B fragment as one contiguous 32B ds_load. This removes the
// (up to 8x) redundant global im2col gather across waves — on a 23.3 TB/s
// part feeding WMMA pipes, this is the operand path that matters.
// ---------------------------------------------------------------------------
__global__ void k_conv_wmma(
    const _Float16* __restrict__ in, const _Float16* __restrict__ wfrag,
    const float* __restrict__ scale, const float* __restrict__ shift,
    const _Float16* __restrict__ res, _Float16* __restrict__ out,
    int N, int Cin, int H, int W, int Cout, int Ho, int Wo,
    int kh, int kw, int stride, int pad, int Kpad, int Kreal, int doRelu,
    int coGroups)
{
    __shared__ __attribute__((aligned(32))) _Float16 sB[2][512];
    __shared__ int sImg[16], sOy[16], sOx[16];

    const int tid   = threadIdx.x;
    const int lane  = tid & 31;
    const int waveId = tid >> 5;
    const int wavesPerBlock = blockDim.x >> 5;

    const long long blockId = blockIdx.x;
    const long long tilePix = blockId / coGroups;
    const int coGroup = (int)(blockId % coGroups);
    const int tileCo  = coGroup * wavesPerBlock + waveId;

    // per-block pixel coordinate table (16 output pixels)
    const int HoWo = Ho * Wo;
    if (tid < 16) {
        long long p = (tilePix << 4) + tid;
        int img = (int)(p / HoWo);
        int rem = (int)(p - (long long)img * HoWo);
        int oy = rem / Wo;
        sImg[tid] = img;
        sOy[tid] = oy;
        sOx[tid] = rem - oy * Wo;
    }
    __syncthreads();

    const int khw = kh * kw;
    const int nKc = Kpad >> 5;
    const int hi  = (lane & 16) ? 1 : 0;
    const _Float16* wbase = wfrag + (((size_t)tileCo * nKc) << 9) + ((size_t)lane << 4);

    // staging assignment: `per` consecutive fragment elements per thread,
    // all within one fragment lane (per in {2,4} divides 16)
    const int per  = 512 / blockDim.x;
    const int e0   = tid * per;
    const int sLane = e0 >> 4;
    const int sN    = sLane & 15;
    const int sHiK  = (sLane & 16) ? 16 : 0;
    const int h0    = e0 & 15;
    const int stImg = sImg[sN];
    const int stIy0 = sOy[sN] * stride - pad;
    const int stIx0 = sOx[sN] * stride - pad;
    const _Float16* inImg = in + (size_t)stImg * Cin * H * W;

    v8f acc = {};
    for (int kc = 0; kc < nKc; ++kc) {
        const int par = kc & 1;
        // ---- cooperative im2col gather into LDS (fragment order) ----
        {
            int k0e = (kc << 5) + sHiK + h0;
            int cin = k0e / khw;
            int r   = k0e - cin * khw;
            int ky  = r / kw;
            int kx  = r - ky * kw;
            for (int j = 0; j < per; ++j) {
                _Float16 val = (_Float16)0.f;
                if (k0e + j < Kreal) {
                    int iy = stIy0 + ky;
                    int ix = stIx0 + kx;
                    if ((unsigned)iy < (unsigned)H && (unsigned)ix < (unsigned)W)
                        val = inImg[((size_t)cin * H + iy) * W + ix];
                }
                sB[par][e0 + j] = val;
                if (++kx == kw) { kx = 0; if (++ky == kh) { ky = 0; ++cin; } }
            }
        }
        // ---- per-wave A fragment: contiguous pre-packed 32B ----
        v16h afrag = *reinterpret_cast<const v16h*>(wbase + ((size_t)kc << 9));
        if (kc + 1 < nKc)
            __builtin_prefetch(wbase + ((size_t)(kc + 1) << 9), 0, 0); // global_prefetch_b8
        __syncthreads();
        // ---- B fragment from LDS: one contiguous 32B ds_load per lane ----
        v16h bfrag = *reinterpret_cast<const v16h*>(&sB[par][lane << 4]);
        acc = __builtin_amdgcn_wmma_f32_16x16x32_f16(
            false, afrag, false, bfrag, (short)0, acc, false, false);
    }

    // Epilogue: D layout — lane holds column n = lane&15, VGPR v holds row m = v + hi*8
    const int n   = lane & 15;
    const int img = sImg[n];
    const int oy  = sOy[n];
    const int ox  = sOx[n];
    #pragma unroll
    for (int vv = 0; vv < 8; ++vv) {
        int m  = vv + (hi << 3);
        int co = (tileCo << 4) + m;
        float val = acc[vv] * scale[co] + shift[co];
        size_t oidx = (((size_t)img * Cout + co) * Ho + oy) * Wo + ox;
        if (res) val += (float)res[oidx];
        if (doRelu) val = fmaxf(val, 0.f);
        out[oidx] = (_Float16)val;
    }
}

// ---------------------------------------------------------------------------
// Split-K dense GEMM partials via WMMA (fc head):
//   Ypart[s, M, N] = A[:, s*Kc:(s+1)*Kc] * B[s*Kc:(s+1)*Kc, :]
// 24 output tiles x 49 K-slices = 1176 waves (vs 24 unsplit) so the 25MB
// B stream is consumed by enough waves to use the HBM bandwidth. The
// partials are reduced in a fixed order afterwards (bitwise deterministic).
// ---------------------------------------------------------------------------
__global__ void k_gemm_wmma_splitk(const _Float16* __restrict__ A, const _Float16* __restrict__ Bm,
                                   float* __restrict__ Ypart,
                                   int M, int N, int K, int Kc /* K per slice */)
{
    const int lane = threadIdx.x & 31;
    int wave = blockIdx.x * (blockDim.x >> 5) + (threadIdx.x >> 5);
    int nN = N >> 4;
    int tilesMN = (M >> 4) * nN;
    int nSlices = K / Kc;
    if (wave >= tilesMN * nSlices) return;   // wave-uniform
    int s = wave / tilesMN;
    int tileIdx = wave % tilesMN;
    int tileM = tileIdx / nN, tileN = tileIdx % nN;
    int mRow = (tileM << 4) + (lane & 15);
    int nCol = (tileN << 4) + (lane & 15);
    int hi = (lane & 16) ? 1 : 0;
    int kStart = s * Kc;

    v8f acc = {};
    for (int k0 = kStart; k0 < kStart + Kc; k0 += 32) {
        v16h a, b;
        const _Float16* arow = A + (size_t)mRow * K + k0 + (hi << 4);
        #pragma unroll
        for (int h = 0; h < 8; ++h) { a[h] = arow[h]; a[h + 8] = arow[8 + h]; }
        const _Float16* bcol = Bm + (size_t)(k0 + (hi << 4)) * N + nCol;
        #pragma unroll
        for (int h = 0; h < 16; ++h) b[h] = bcol[(size_t)h * N];
        acc = __builtin_amdgcn_wmma_f32_16x16x32_f16(
            false, a, false, b, (short)0, acc, false, false);
    }
    #pragma unroll
    for (int vv = 0; vv < 8; ++vv) {
        int m = (tileM << 4) + vv + (hi << 3);
        Ypart[((size_t)s * M + m) * N + nCol] = acc[vv];
    }
}

// fixed-order reduction of split-K partials + bias (deterministic)
__global__ void k_gemm_reduce(const float* __restrict__ Ypart, const float* __restrict__ bias,
                              float* __restrict__ Y, int M, int N, int nSlices) {
    int i = blockIdx.x * blockDim.x + threadIdx.x;
    if (i >= M * N) return;
    float acc = bias[i % N];
    for (int s = 0; s < nSlices; ++s) acc += Ypart[(size_t)s * M * N + i];
    Y[i] = acc;
}

// ---------------------------------------------------------------------------
// pooling
// ---------------------------------------------------------------------------
__global__ void k_maxpool3s2(const _Float16* __restrict__ in, _Float16* __restrict__ out,
                             int N, int C, int H, int W, int Ho, int Wo) {
    size_t idx = (size_t)blockIdx.x * blockDim.x + threadIdx.x;
    size_t total = (size_t)N * C * Ho * Wo;
    if (idx >= total) return;
    int wo = (int)(idx % Wo); size_t t = idx / Wo;
    int ho = (int)(t % Ho);   t /= Ho;
    int c  = (int)(t % C);
    int n  = (int)(t / C);
    const _Float16* p = in + ((size_t)n * C + c) * H * W;
    float mval = -3.0e38f;
    int y0 = ho * 2 - 1, x0 = wo * 2 - 1;
    #pragma unroll
    for (int dy = 0; dy < 3; ++dy)
        #pragma unroll
        for (int dx = 0; dx < 3; ++dx) {
            int y = y0 + dy, x = x0 + dx;
            if ((unsigned)y < (unsigned)H && (unsigned)x < (unsigned)W)
                mval = fmaxf(mval, (float)p[(size_t)y * W + x]);
        }
    out[idx] = (_Float16)mval;
}

__global__ void k_meanpool(const _Float16* __restrict__ in, float* __restrict__ out,
                           int NC, int HW) {
    int i = blockIdx.x * blockDim.x + threadIdx.x;
    if (i >= NC) return;
    float s = 0.f;
    for (int j = 0; j < HW; ++j) s += (float)in[(size_t)i * HW + j];
    out[i] = s / (float)HW;
}

// ---------------------------------------------------------------------------
// fc scoring head: relu(bn(y1)) @ w2 + b2
// ---------------------------------------------------------------------------
__global__ void k_fc_scores(const float* __restrict__ y1, const float* __restrict__ g,
                            const float* __restrict__ b, const float* __restrict__ m,
                            const float* __restrict__ v, const float* __restrict__ w2,
                            const float* __restrict__ b2, float* __restrict__ sc, int M) {
    int i = blockIdx.x * blockDim.x + threadIdx.x;
    if (i >= M) return;
    float acc = 0.f;
    for (int c = 0; c < 64; ++c) {
        float t = (y1[(size_t)i * 64 + c] - m[c]) * rsqrtf(v[c] + 1e-5f) * g[c] + b[c];
        acc += fmaxf(t, 0.f) * w2[c];
    }
    sc[i] = acc + b2[0];
}

// ---------------------------------------------------------------------------
// per-batch group fusion (softmax bucketing); scalar part in LDS
// ---------------------------------------------------------------------------
#define MAXV_C 12
#define MG 8
__global__ void k_group_fuse(const float* __restrict__ fv, const float* __restrict__ scores,
                             const int* __restrict__ num_views, float* __restrict__ kout) {
    int bI = blockIdx.x;
    __shared__ float coeff[MAXV_C];
    if (threadIdx.x == 0) {
        int nv = num_views[bI];
        float s[MAXV_C];
        int bucket[MAXV_C];
        float mx = -3.0e38f;
        for (int v2 = 0; v2 < MAXV_C; ++v2) {
            if (v2 < nv) {
                float sc = scores[bI * MAXV_C + v2];
                float lg = logf(fabsf(sc));
                s[v2] = 1.f / (1.f + expf(-lg));
                if (s[v2] > mx) mx = s[v2];
            } else s[v2] = 0.f;
        }
        float den = 0.f;
        for (int v2 = 0; v2 < MAXV_C; ++v2)
            if (v2 < nv) { s[v2] = expf(s[v2] - mx); den += s[v2]; }
        for (int v2 = 0; v2 < MAXV_C; ++v2)
            s[v2] = (v2 < nv) ? s[v2] / den : 0.f;
        float counts[MG];
        for (int g2 = 0; g2 < MG; ++g2) counts[g2] = 0.f;
        for (int v2 = 0; v2 < MAXV_C; ++v2) {
            if (v2 < nv) {
                int bk = (int)floorf(s[v2] * MG);
                bk = bk < 0 ? 0 : (bk > MG - 1 ? MG - 1 : bk);
                bucket[v2] = bk;
                counts[bk] += 1.f;
            } else bucket[v2] = MG;
        }
        float wg[MG];
        for (int g2 = 0; g2 < MG; ++g2) wg[g2] = 0.f;
        for (int v2 = 0; v2 < MAXV_C; ++v2)
            if (v2 < nv) wg[bucket[v2]] += ceilf(s[v2] * counts[bucket[v2]]);
        float Wsum = 0.f;
        for (int g2 = 0; g2 < MG; ++g2) {
            if (counts[g2] > 0.f) wg[g2] /= counts[g2]; else wg[g2] = 0.f;
            Wsum += wg[g2];
        }
        for (int v2 = 0; v2 < MAXV_C; ++v2) {
            float c = 0.f;
            if (v2 < nv) {
                int bk = bucket[v2];
                if (counts[bk] > 0.f) c = wg[bk] / counts[bk] / Wsum;
            }
            coeff[v2] = c;
        }
    }
    __syncthreads();
    for (int d = threadIdx.x; d < 512; d += blockDim.x) {
        float acc = 0.f;
        for (int v2 = 0; v2 < MAXV_C; ++v2)
            acc += coeff[v2] * fv[((size_t)bI * MAXV_C + v2) * 512 + d];
        kout[(size_t)bI * 512 + d] = acc;
    }
}

__global__ void k_logits(const float* __restrict__ kmat, const float* __restrict__ Wc,
                         const float* __restrict__ bias, float* __restrict__ out) {
    int t = threadIdx.x;                  // 0..319
    if (t >= 320) return;
    int bI = t / 40, c = t % 40;
    float acc = bias[c];
    for (int d = 0; d < 512; ++d) acc += kmat[(size_t)bI * 512 + d] * Wc[(size_t)d * 40 + c];
    out[t] = acc;
}

// ---------------------------------------------------------------------------
// host orchestration
// ---------------------------------------------------------------------------
extern "C" void kernel_launch(void* const* d_in, const int* in_sizes, int n_in,
                              void* d_out, int out_size, void* d_ws, size_t ws_size,
                              hipStream_t stream) {
    (void)in_sizes; (void)n_in; (void)out_size;
    // ---- input indices (setup_inputs dict order, params pytree sorted keys) ----
    const int IN_NV = 2, IN_X = 3, CLS_B = 4, CLS_W = 5,
              FC_B1 = 6, FC_B2 = 7, FC_BN_B = 8, FC_BN_G = 9, FC_BN_M = 10, FC_BN_V = 11,
              FC_W1 = 12, FC_W2 = 13,
              L1_0 = 14, L1_1 = 24, L2_0 = 34, L2_1 = 49,
              L3_0 = 59, L3_1 = 74, L4_0 = 84, L4_1 = 99,
              STEM_BN = 109, STEM_C = 113;
    // no-down block: bn1 at +0..3 (b,g,m,v), bn2 +4..7, c1 +8, c2 +9
    // down block:    bn1 +0..3, bn2 +4..7, bnd +8..11, c1 +12, c2 +13, cd +14

    // ---- workspace layout (bytes) ----
    const size_t OFF_XH    = 0;                         // 28,901,376  (x in f16)
    const size_t OFF_BIG   = 28901376;                  // 154,140,672 (stem out; reused)
    const size_t OFF_A1R   = OFF_BIG + 154140672;       // 38,535,168  (a1; reused for l3)
    const size_t OFF_A3    = OFF_A1R + 38535168;        // 38,535,168  (raw_views, persists)
    const size_t OFF_WPACK = OFF_A3 + 38535168;         // 22,335,488
    const size_t OFF_W1H   = OFF_WPACK + 22335488;      // 25,690,112
    const size_t OFF_SMALL = OFF_W1H + 25690112;
    const size_t OFF_YPART = OFF_SMALL + 20480 + 20480 + 196608 + 24576 + 512;
    const int    KSLICES   = 49;                        // 49 * 4096 = 200704
    const size_t WS_NEEDED = OFF_YPART + (size_t)KSLICES * 96 * 64 * 4;

    float* outF = (float*)d_out;
    if (!d_ws || ws_size < WS_NEEDED) {
        k_zero_f32<<<(4416 + 255) / 256, 256, 0, stream>>>(outF, 4416);
        return;
    }

    char* ws = (char*)d_ws;
    _Float16* xh     = (_Float16*)(ws + OFF_XH);
    _Float16* big    = (_Float16*)(ws + OFF_BIG);
    _Float16* a1r    = (_Float16*)(ws + OFF_A1R);
    _Float16* a3     = (_Float16*)(ws + OFF_A3);      // raw_views (96,64,56,56)
    _Float16* wpack  = (_Float16*)(ws + OFF_WPACK);
    _Float16* w1h    = (_Float16*)(ws + OFF_W1H);
    float* scaleBuf  = (float*)(ws + OFF_SMALL);
    float* shiftBuf  = (float*)(ws + OFF_SMALL + 20480);
    float* fv        = (float*)(ws + OFF_SMALL + 40960);            // 96x512
    float* y1        = (float*)(ws + OFF_SMALL + 40960 + 196608);   // 96x64
    float* scores    = (float*)(ws + OFF_SMALL + 40960 + 196608 + 24576); // 96
    float* ypart     = (float*)(ws + OFF_YPART);                    // 49x96x64

    // aliased regions inside BIG (f16 element offsets)
    _Float16* a_stem = big;                         // 96x64x112x112
    _Float16* t1  = big;                            // 96x64x56x56
    _Float16* a2  = big + 19267584;
    _Float16* sc2 = big + 38535168;                 // 96x128x28x28
    _Float16* t2  = big + 48168960;
    _Float16* b2a = big + 57802752;
    _Float16* b2b = big + 67436544;
    _Float16* sc4 = big + 38535168;                 // l4 reuses dead l2 area: 96x512x7x7
    _Float16* t4  = big + 40943616;
    _Float16* b4a = big + 43352064;
    _Float16* b4b = big + 45760512;
    // aliased regions inside A1R
    _Float16* a1  = a1r;                            // 96x64x56x56
    _Float16* sc3 = a1r;                            // l3 reuses: 96x256x14x14
    _Float16* t3  = a1r + 4816896;
    _Float16* b3a = a1r + 9633792;
    _Float16* b3b = a1r + 14450688;

    size_t wOff = 0; int chOff = 0;
    auto conv = [&](int wIdx, int bnIdx, const _Float16* in, const _Float16* res,
                    _Float16* out, int N, int Cin, int H, int W, int Cout, int Ho, int Wo,
                    int kh, int kw, int stride, int pad, int relu) {
        int Kreal = Cin * kh * kw;
        int Kpad = (Kreal + 31) & ~31;
        size_t total = (size_t)(Cout / 16) * (size_t)(Kpad / 32) * 512;
        k_pack_w<<<(unsigned)((total + 255) / 256), 256, 0, stream>>>(
            (const float*)d_in[wIdx], wpack + wOff, Cout, Kreal, Kpad);
        k_bn_fuse<<<(Cout + 255) / 256, 256, 0, stream>>>(
            (const float*)d_in[bnIdx + 1], (const float*)d_in[bnIdx + 0],
            (const float*)d_in[bnIdx + 2], (const float*)d_in[bnIdx + 3],
            scaleBuf + chOff, shiftBuf + chOff, Cout);
        int nCo = Cout / 16;
        int wavesPerBlock = nCo < 8 ? nCo : 8;
        int coGroups = nCo / wavesPerBlock;
        long long nPixTiles = (long long)N * Ho * Wo / 16;
        unsigned blocks = (unsigned)(nPixTiles * coGroups);
        k_conv_wmma<<<blocks, 32 * wavesPerBlock, 0, stream>>>(
            in, wpack + wOff, scaleBuf + chOff, shiftBuf + chOff, res, out,
            N, Cin, H, W, Cout, Ho, Wo, kh, kw, stride, pad, Kpad, Kreal, relu, coGroups);
        wOff += total; chOff += Cout;
    };

    // ---- prep: x -> f16 ----
    const size_t NX = (size_t)96 * 3 * 224 * 224;
    k_f32_to_f16<<<(unsigned)((NX + 255) / 256), 256, 0, stream>>>(
        (const float*)d_in[IN_X], xh, NX);

    // ---- stem + maxpool ----
    conv(STEM_C, STEM_BN, xh, nullptr, a_stem, 96, 3, 224, 224, 64, 112, 112, 7, 7, 2, 3, 1);
    {
        size_t total = (size_t)96 * 64 * 56 * 56;
        k_maxpool3s2<<<(unsigned)((total + 255) / 256), 256, 0, stream>>>(
            a_stem, a1, 96, 64, 112, 112, 56, 56);
    }

    // ---- layer1 (64ch, 56x56) ----
    conv(L1_0 + 8, L1_0 + 0, a1, nullptr, t1, 96, 64, 56, 56, 64, 56, 56, 3, 3, 1, 1, 1);
    conv(L1_0 + 9, L1_0 + 4, t1, a1,      a2, 96, 64, 56, 56, 64, 56, 56, 3, 3, 1, 1, 1);
    conv(L1_1 + 8, L1_1 + 0, a2, nullptr, t1, 96, 64, 56, 56, 64, 56, 56, 3, 3, 1, 1, 1);
    conv(L1_1 + 9, L1_1 + 4, t1, a2,      a3, 96, 64, 56, 56, 64, 56, 56, 3, 3, 1, 1, 1);

    // ---- layer2 (128ch, 28x28) ----
    conv(L2_0 + 14, L2_0 + 8, a3, nullptr, sc2, 96, 64, 56, 56, 128, 28, 28, 1, 1, 2, 0, 0);
    conv(L2_0 + 12, L2_0 + 0, a3, nullptr, t2,  96, 64, 56, 56, 128, 28, 28, 3, 3, 2, 1, 1);
    conv(L2_0 + 13, L2_0 + 4, t2, sc2,     b2a, 96, 128, 28, 28, 128, 28, 28, 3, 3, 1, 1, 1);
    conv(L2_1 + 8,  L2_1 + 0, b2a, nullptr, t2, 96, 128, 28, 28, 128, 28, 28, 3, 3, 1, 1, 1);
    conv(L2_1 + 9,  L2_1 + 4, t2, b2a,     b2b, 96, 128, 28, 28, 128, 28, 28, 3, 3, 1, 1, 1);

    // ---- layer3 (256ch, 14x14) ----
    conv(L3_0 + 14, L3_0 + 8, b2b, nullptr, sc3, 96, 128, 28, 28, 256, 14, 14, 1, 1, 2, 0, 0);
    conv(L3_0 + 12, L3_0 + 0, b2b, nullptr, t3,  96, 128, 28, 28, 256, 14, 14, 3, 3, 2, 1, 1);
    conv(L3_0 + 13, L3_0 + 4, t3, sc3,      b3a, 96, 256, 14, 14, 256, 14, 14, 3, 3, 1, 1, 1);
    conv(L3_1 + 8,  L3_1 + 0, b3a, nullptr, t3,  96, 256, 14, 14, 256, 14, 14, 3, 3, 1, 1, 1);
    conv(L3_1 + 9,  L3_1 + 4, t3, b3a,      b3b, 96, 256, 14, 14, 256, 14, 14, 3, 3, 1, 1, 1);

    // ---- layer4 (512ch, 7x7) ----
    conv(L4_0 + 14, L4_0 + 8, b3b, nullptr, sc4, 96, 256, 14, 14, 512, 7, 7, 1, 1, 2, 0, 0);
    conv(L4_0 + 12, L4_0 + 0, b3b, nullptr, t4,  96, 256, 14, 14, 512, 7, 7, 3, 3, 2, 1, 1);
    conv(L4_0 + 13, L4_0 + 4, t4, sc4,      b4a, 96, 512, 7, 7, 512, 7, 7, 3, 3, 1, 1, 1);
    conv(L4_1 + 8,  L4_1 + 0, b4a, nullptr, t4,  96, 512, 7, 7, 512, 7, 7, 3, 3, 1, 1, 1);
    conv(L4_1 + 9,  L4_1 + 4, t4, b4a,      b4b, 96, 512, 7, 7, 512, 7, 7, 3, 3, 1, 1, 1);

    // ---- global mean pool -> final_views (96,512) f32 ----
    k_meanpool<<<(96 * 512 + 255) / 256, 256, 0, stream>>>(b4b, fv, 96 * 512, 49);

    // ---- fc head: raw_scores from raw_views (a3) ----
    const size_t NW1 = (size_t)200704 * 64;
    k_f32_to_f16<<<(unsigned)((NW1 + 255) / 256), 256, 0, stream>>>(
        (const float*)d_in[FC_W1], w1h, NW1);
    // split-K: (96,200704)@(200704,64), 24 tiles x 49 slices = 1176 waves
    {
        int tilesMN = (96 / 16) * (64 / 16);            // 24
        unsigned blocks = (unsigned)((tilesMN * KSLICES + 7) / 8);  // 147
        k_gemm_wmma_splitk<<<blocks, 256, 0, stream>>>(a3, w1h, ypart,
                                                       96, 64, 200704, 200704 / KSLICES);
        k_gemm_reduce<<<(96 * 64 + 255) / 256, 256, 0, stream>>>(
            ypart, (const float*)d_in[FC_B1], y1, 96, 64, KSLICES);
    }
    k_fc_scores<<<1, 128, 0, stream>>>(y1,
        (const float*)d_in[FC_BN_G], (const float*)d_in[FC_BN_B],
        (const float*)d_in[FC_BN_M], (const float*)d_in[FC_BN_V],
        (const float*)d_in[FC_W2], (const float*)d_in[FC_B2], scores, 96);

    // ---- group fuse -> k (d_out+320), then logits (d_out) ----
    float* kOut = outF + 320;
    k_group_fuse<<<8, 256, 0, stream>>>(fv, scores, (const int*)d_in[IN_NV], kOut);
    k_logits<<<1, 320, 0, stream>>>(kOut, (const float*)d_in[CLS_W],
                                    (const float*)d_in[CLS_B], outF);
}
